// quantized_conv_72258529788396
// MI455X (gfx1250) — 8-bit matrix operand order repaired and checked by reading, compiled, not run on hardware
//
#include <hip/hip_runtime.h>
#include <stdint.h>

typedef __attribute__((ext_vector_type(8))) int v8i;

namespace {

constexpr int kB = 2, kCin = 128, kP = 1024;
constexpr int kCout = 128, kF = 1152;
constexpr int kNsub = 9;
constexpr int kM = kB * kP * 8;
constexpr int kN = kCout * 8;
constexpr int kMtiles = kM / 16;
constexpr int kNtiles = kN / 16;
constexpr int kKtiles = kF / 64;
constexpr int kOutElems = kB * kCout * kP;

constexpr size_t QX_OFF    = 0;
constexpr size_t NORM_OFF  = 262144;
constexpr size_t VMAX_OFF  = 262656;
constexpr size_t ACC_OFF   = 263168;
constexpr size_t APACK_OFF = 1311744;
constexpr size_t BPOS_OFF  = APACK_OFF + (size_t)kMtiles * kKtiles * 1024;
constexpr size_t BNEG_OFF  = BPOS_OFF  + (size_t)kNtiles * kKtiles * 1024;

__device__ __forceinline__ v8i load_v8i(const uint8_t* p) {
  const int4* q = reinterpret_cast<const int4*>(p);
  int4 a = q[0], b = q[1];
  v8i r;
  r[0] = a.x; r[1] = a.y; r[2] = a.z; r[3] = a.w;
  r[4] = b.x; r[5] = b.y; r[6] = b.z; r[7] = b.w;
  return r;
}

__device__ __forceinline__ v8i wmma_iu8(v8i a, v8i b, v8i c) {
  return __builtin_amdgcn_wmma_i32_16x16x64_iu8(false, a, false, b, c, false, false);
}

}


__global__ void k_init(float* __restrict__ acc, int* __restrict__ vmax) {
  int i = blockIdx.x * blockDim.x + threadIdx.x;
  if (i < kOutElems) acc[i] = 0.0f;
  if (i < 18) vmax[i] = 0;
}

__global__ void k_quant_x(const float* __restrict__ x, uint8_t* __restrict__ qx) {
  int i = blockIdx.x * blockDim.x + threadIdx.x;
  if (i < kB * kCin * 32 * 32) {
    float v = rintf(x[i] * 256.0f);
    v = fminf(fmaxf(v, 0.0f), 255.0f);
    qx[i] = (uint8_t)v;
  }
}

__global__ __launch_bounds__(256) void k_norm(const float* __restrict__ w,
                                              float* __restrict__ norm) {
  __shared__ float red[256];
  int o = blockIdx.x;
  int tid = threadIdx.x;
  float m = 0.0f;
  for (int i = tid; i < kF; i += 256) m = fmaxf(m, fabsf(w[o * kF + i]));
  red[tid] = m;
  __syncthreads();
  for (int s = 128; s > 0; s >>= 1) {
    if (tid < s) red[tid] = fmaxf(red[tid], red[tid + s]);
    __syncthreads();
  }
  if (tid == 0) norm[o] = fmaxf(red[0], 1e-20f);
}


__global__ void k_pack_a(const uint8_t* __restrict__ qx, uint32_t* __restrict__ apack) {
  int idx = blockIdx.x * blockDim.x + threadIdx.x;
  if (idx >= kMtiles * kKtiles * 32 * 8) return;
  int j    = idx & 7;
  int lane = (idx >> 3) & 31;
  int blk  = idx >> 8;
  int kt   = blk % kKtiles;
  int mtile = blk / kKtiles;
  int baseK = ((lane >> 4) << 3) | ((j & 1) << 2) | (((j >> 1) & 1) << 4) | ((j >> 2) << 5);
  int m = (mtile << 4) + (lane & 15);
  int b = m >> 13;
  int p = (m >> 3) & 1023;
  int s = m & 7;
  int oh = p >> 5, ow = p & 31;
  uint32_t word = 0;
  for (int by = 0; by < 4; ++by) {
    int f = kt * 64 + baseK + by;
    int c = f / 9, r = f - c * 9;
    int kh = r / 3, kw = r - kh * 3;
    int ih = oh + kh - 1, iw = ow + kw - 1;
    uint32_t bit = 0;
    if (ih >= 0 && ih < 32 && iw >= 0 && iw < 32) {
      uint8_t q = qx[((b * kCin + c) * 32 + ih) * 32 + iw];
      bit = (q >> s) & 1u;
    }
    word |= bit << (8 * by);
  }
  apack[idx] = word;
}

__global__ void k_pack_b(const float* __restrict__ w, const float* __restrict__ norm,
                         uint32_t* __restrict__ bpos, uint32_t* __restrict__ bneg) {
  int idx = blockIdx.x * blockDim.x + threadIdx.x;
  if (idx >= kNtiles * kKtiles * 32 * 8) return;
  int j    = idx & 7;
  int lane = (idx >> 3) & 31;
  int blk  = idx >> 8;
  int kt   = blk % kKtiles;
  int ntile = blk / kKtiles;
  int baseK = ((lane >> 4) << 3) | ((j & 1) << 2) | (((j >> 1) & 1) << 4) | ((j >> 2) << 5);
  int n = (ntile << 4) + (lane & 15);
  int o = n >> 3, t = n & 7;
  float nm = norm[o];
  uint32_t wp = 0, wn = 0;
  for (int by = 0; by < 4; ++by) {
    int f = kt * 64 + baseK + by;
    int c = f / 9, r = f - c * 9;
    int kh = r / 3, kw = r - kh * 3;
    float wv = w[((o * kCin + c) * 3 + kh) * 3 + kw];
    int wi = (int)rintf(wv / nm * 255.0f);
    uint32_t pb = (wi > 0) ? (uint32_t)((wi >> t) & 1) : 0u;
    uint32_t nb = (wi < 0) ? (uint32_t)(((-wi) >> t) & 1) : 0u;
    wp |= pb << (8 * by);
    wn |= nb << (8 * by);
  }
  bpos[idx] = wp;
  bneg[idx] = wn;
}


__global__ __launch_bounds__(256) void k_gemm_max(const uint8_t* __restrict__ apack,
                                                  const uint8_t* __restrict__ bpos,
                                                  const uint8_t* __restrict__ bneg,
                                                  int* __restrict__ vmaxp,
                                                  int* __restrict__ vmaxn) {
  int mtile = blockIdx.x;
  int chunk = blockIdx.y;
  int wave = threadIdx.x >> 5;
  int lane = threadIdx.x & 31;
  int kt0 = chunk * 2, kt1 = kt0 + 1;

  v8i a0 = load_v8i(apack + ((size_t)(mtile * kKtiles + kt0)) * 1024 + lane * 32);
  v8i a1 = load_v8i(apack + ((size_t)(mtile * kKtiles + kt1)) * 1024 + lane * 32);

  int lmp = 0, lmn = 0;
  for (int q = 0; q < 8; ++q) {
    int ntile = wave * 8 + q;
    const uint8_t* pb0 = bpos + ((size_t)(ntile * kKtiles + kt0)) * 1024 + lane * 32;
    const uint8_t* nb0 = bneg + ((size_t)(ntile * kKtiles + kt0)) * 1024 + lane * 32;
    if (q < 7) {
      __builtin_prefetch(pb0 + (size_t)kKtiles * 1024, 0, 1);
      __builtin_prefetch(nb0 + (size_t)kKtiles * 1024, 0, 1);
    }
    v8i bp0 = load_v8i(pb0);
    v8i bp1 = load_v8i(pb0 + 1024);
    v8i bn0 = load_v8i(nb0);
    v8i bn1 = load_v8i(nb0 + 1024);
    v8i zero = {0, 0, 0, 0, 0, 0, 0, 0};
    v8i accp = wmma_iu8(a0, bp0, zero);
    accp     = wmma_iu8(a1, bp1, accp);
    v8i accn = wmma_iu8(a0, bn0, zero);
    accn     = wmma_iu8(a1, bn1, accn);
#pragma unroll
    for (int jj = 0; jj < 8; ++jj) {
      lmp = max(lmp, accp[jj]);
      lmn = max(lmn, accn[jj]);
    }
  }
  __shared__ int sp, sn;
  if (threadIdx.x == 0) { sp = 0; sn = 0; }
  __syncthreads();
  atomicMax(&sp, lmp);
  atomicMax(&sn, lmn);
  __syncthreads();
  if (threadIdx.x == 0) {
    atomicMax(&vmaxp[chunk], sp);
    atomicMax(&vmaxn[chunk], sn);
  }
}


__global__ __launch_bounds__(256) void k_gemm_adc(const uint8_t* __restrict__ apack,
                                                  const uint8_t* __restrict__ bpos,
                                                  const uint8_t* __restrict__ bneg,
                                                  const int* __restrict__ vmaxp,
                                                  const int* __restrict__ vmaxn,
                                                  float* __restrict__ acc) {
  int mtile = blockIdx.x;
  int chunk = blockIdx.y;
  int wave = threadIdx.x >> 5;
  int lane = threadIdx.x & 31;
  int kt0 = chunk * 2, kt1 = kt0 + 1;

  float vp = fmaxf((float)vmaxp[chunk], 1e-6f);
  float vn = fmaxf((float)vmaxn[chunk], 1e-6f);
  float stepp = vp * (1.0f / 15.0f);
  float stepn = vn * (1.0f / 15.0f);
  float rstepp = 1.0f / stepp;
  float rstepn = 1.0f / stepn;

  v8i a0 = load_v8i(apack + ((size_t)(mtile * kKtiles + kt0)) * 1024 + lane * 32);
  v8i a1 = load_v8i(apack + ((size_t)(mtile * kKtiles + kt1)) * 1024 + lane * 32);

  int b = mtile >> 9;
  int p = (((mtile << 1) | (lane >> 4)) & 1023);
  int Nloc = lane & 15;
  float tscale = (float)(1 << (Nloc & 7));

  for (int q = 0; q < 8; ++q) {
    int ntile = wave * 8 + q;
    const uint8_t* pb0 = bpos + ((size_t)(ntile * kKtiles + kt0)) * 1024 + lane * 32;
    const uint8_t* nb0 = bneg + ((size_t)(ntile * kKtiles + kt0)) * 1024 + lane * 32;
    if (q < 7) {
      __builtin_prefetch(pb0 + (size_t)kKtiles * 1024, 0, 1);
      __builtin_prefetch(nb0 + (size_t)kKtiles * 1024, 0, 1);
    }
    v8i bp0 = load_v8i(pb0);
    v8i bp1 = load_v8i(pb0 + 1024);
    v8i bn0 = load_v8i(nb0);
    v8i bn1 = load_v8i(nb0 + 1024);
    v8i zero = {0, 0, 0, 0, 0, 0, 0, 0};
    v8i accp = wmma_iu8(a0, bp0, zero);
    accp     = wmma_iu8(a1, bp1, accp);
    v8i accn = wmma_iu8(a0, bn0, zero);
    accn     = wmma_iu8(a1, bn1, accn);

    float ssum = 0.0f;
#pragma unroll
    for (int jj = 0; jj < 8; ++jj) {
      float ap = fminf(fmaxf(rintf((float)accp[jj] * rstepp), 0.0f), 15.0f) * stepp;
      float an = fminf(fmaxf(rintf((float)accn[jj] * rstepn), 0.0f), 15.0f) * stepn;
      ssum += (ap - an) * (float)(1 << jj);
    }
    float v = ssum * tscale;

    v += __shfl_xor(v, 1, 32);
    v += __shfl_xor(v, 2, 32);
    v += __shfl_xor(v, 4, 32);

    int o = (ntile << 1) + (Nloc >> 3);
    if ((lane & 7) == 0) {
      atomicAdd(&acc[((b << 7) + o) * kP + p], v);
    }
  }
}


__global__ void k_final(const float* __restrict__ acc, const float* __restrict__ norm,
                        float* __restrict__ out, int out_size) {
  int i = blockIdx.x * blockDim.x + threadIdx.x;
  if (i < kOutElems) {
    int o = (i >> 10) & 127;
    out[i] = acc[i] * norm[o] * (1.0f / (255.0f * 256.0f));
  }
  if (i == 0 && out_size > kOutElems) out[kOutElems] = 0.0f;
}

extern "C" void kernel_launch(void* const* d_in, const int* in_sizes, int n_in,
                              void* d_out, int out_size, void* d_ws, size_t ws_size,
                              hipStream_t stream) {
  (void)in_sizes; (void)n_in; (void)ws_size;
  const float* x = (const float*)d_in[0];
  const float* w = (const float*)d_in[1];
  char* ws = (char*)d_ws;

  uint8_t*  qx    = (uint8_t*) (ws + QX_OFF);
  float*    norm  = (float*)   (ws + NORM_OFF);
  int*      vmaxp = (int*)     (ws + VMAX_OFF);
  int*      vmaxn = vmaxp + kNsub;
  float*    acc   = (float*)   (ws + ACC_OFF);
  uint32_t* apack = (uint32_t*)(ws + APACK_OFF);
  uint32_t* bpos  = (uint32_t*)(ws + BPOS_OFF);
  uint32_t* bneg  = (uint32_t*)(ws + BNEG_OFF);

  k_init<<<(kOutElems + 255) / 256, 256, 0, stream>>>(acc, vmaxp);
  k_quant_x<<<(kOutElems + 255) / 256, 256, 0, stream>>>(x, qx);
  k_norm<<<kCout, 256, 0, stream>>>(w, norm);

  int a_words = kMtiles * kKtiles * 32 * 8;
  int b_words = kNtiles * kKtiles * 32 * 8;
  k_pack_a<<<(a_words + 255) / 256, 256, 0, stream>>>(qx, apack);
  k_pack_b<<<(b_words + 255) / 256, 256, 0, stream>>>(w, norm, bpos, bneg);

  dim3 ggrid(kMtiles, kNsub);
  k_gemm_max<<<ggrid, 256, 0, stream>>>((const uint8_t*)apack, (const uint8_t*)bpos,
                                        (const uint8_t*)bneg, vmaxp, vmaxn);
  k_gemm_adc<<<ggrid, 256, 0, stream>>>((const uint8_t*)apack, (const uint8_t*)bpos,
                                        (const uint8_t*)bneg, vmaxp, vmaxn, acc);

  k_final<<<(kOutElems + 255) / 256, 256, 0, stream>>>(acc, norm, (float*)d_out, out_size);
}
